// ProteinFeatures_24721831756678
// MI455X (gfx1250) — matrix operand order repaired and checked by reading, not compiled, not run on hardware
//
#include <hip/hip_runtime.h>
#include <hip/hip_bf16.h>

typedef __attribute__((ext_vector_type(16))) _Float16 v16h;
typedef __attribute__((ext_vector_type(8)))  float    v8f;

#define N_RES 4096
#define KNN 48
#define EDIM 128
#define DFEAT 416
#define MAXREL 16

// ---------------------------------------------------------------------------
// Kernel 1: backbone construction  bb[i] = {n, ca, c, o, cb} (5 atoms x 3)
// ---------------------------------------------------------------------------
__global__ void pf_prep_bb(const float* __restrict__ coords, float* __restrict__ bb) {
    int i = blockIdx.x * blockDim.x + threadIdx.x;
    if (i >= N_RES) return;
    const float* p = coords + (size_t)i * 12;
    float nx = p[0], ny = p[1], nz = p[2];
    float cax = p[3], cay = p[4], caz = p[5];
    float cx = p[6], cy = p[7], cz = p[8];
    float ox = p[9], oy = p[10], oz = p[11];
    float bx = cax - nx, by = cay - ny, bz = caz - nz;
    float ccx = cx - cax, ccy = cy - cay, ccz = cz - caz;
    float ax = by * ccz - bz * ccy;
    float ay = bz * ccx - bx * ccz;
    float az = bx * ccy - by * ccx;
    float cbx = -0.58273431f * ax + 0.56802827f * bx - 0.54067466f * ccx + cax;
    float cby = -0.58273431f * ay + 0.56802827f * by - 0.54067466f * ccy + cay;
    float cbz = -0.58273431f * az + 0.56802827f * bz - 0.54067466f * ccz + caz;
    float* o = bb + (size_t)i * 15;
    o[0] = nx;  o[1] = ny;  o[2] = nz;
    o[3] = cax; o[4] = cay; o[5] = caz;
    o[6] = cx;  o[7] = cy;  o[8] = cz;
    o[9] = ox;  o[10] = oy; o[11] = oz;
    o[12] = cbx; o[13] = cby; o[14] = cbz;
}

// ---------------------------------------------------------------------------
// Kernel 2: f32 -> f16 weight prep, swizzled into per-lane WMMA B-fragment
// order: element t = ((kk*8 + wave)*32 + lane)*16 + h holds
//   B[kdim][col]  with col = wave*16 + (lane&15),
//                      kdim = kk*32 + (lane&16 ? 16 : 0) + h
// so each lane's 16 halves for one WMMA step are one aligned 32-byte load.
// ---------------------------------------------------------------------------
__global__ void pf_prep_w(const float* __restrict__ w_e_w,
                          const float* __restrict__ proj_w,
                          _Float16* __restrict__ w_e_t,
                          _Float16* __restrict__ proj_t) {
    int t = blockIdx.x * blockDim.x + threadIdx.x;
    if (t < DFEAT * EDIM) {
        int h    = t & 15;
        int lane = (t >> 4) & 31;
        int wave = (t >> 9) & 7;
        int kk   = t >> 12;
        int col  = wave * 16 + (lane & 15);
        int kd   = kk * 32 + ((lane & 16) ? 16 : 0) + h;
        w_e_t[t] = (_Float16)w_e_w[col * DFEAT + kd];   // w_e_w[e][d]
    } else if (t < DFEAT * EDIM + EDIM * EDIM) {
        int q    = t - DFEAT * EDIM;
        int h    = q & 15;
        int lane = (q >> 4) & 31;
        int wave = (q >> 9) & 7;
        int kk   = q >> 12;
        int col  = wave * 16 + (lane & 15);
        int kd   = kk * 32 + ((lane & 16) ? 16 : 0) + h;
        proj_t[q] = (_Float16)proj_w[col * EDIM + kd];  // proj_w[f][e]
    }
}

// ---------------------------------------------------------------------------
// Kernel 3: K nearest neighbors by CA distance (matches top_k(-dist) order,
// lowest-index tie break). One 256-thread block per residue.
// ---------------------------------------------------------------------------
__global__ void pf_topk(const float* __restrict__ coords,
                        const float* __restrict__ mask,
                        int* __restrict__ idxout) {
    __shared__ float d2s[N_RES];
    __shared__ float rv[256];
    __shared__ int   ri[256];
    const int tid = threadIdx.x;
    const int i = blockIdx.x;
    const float cax = coords[(size_t)i * 12 + 3];
    const float cay = coords[(size_t)i * 12 + 4];
    const float caz = coords[(size_t)i * 12 + 5];
    const float mi = mask[i];
    for (int j = tid; j < N_RES; j += 256) {
        float dx = cax - coords[(size_t)j * 12 + 3];
        float dy = cay - coords[(size_t)j * 12 + 4];
        float dz = caz - coords[(size_t)j * 12 + 5];
        float d2 = dx * dx + dy * dy + dz * dz;
        if (mi == 0.0f || mask[j] == 0.0f) d2 = 3.38e38f;
        d2s[j] = d2;
    }
    __syncthreads();
    for (int it = 0; it < KNN; ++it) {
        float best = 3.3e38f;
        int bi = N_RES;
        for (int j = tid; j < N_RES; j += 256) {
            float v = d2s[j];
            if (v < best || (v == best && j < bi)) { best = v; bi = j; }
        }
        rv[tid] = best; ri[tid] = bi;
        __syncthreads();
        for (int s = 128; s > 0; s >>= 1) {
            if (tid < s) {
                if (rv[tid + s] < rv[tid] ||
                    (rv[tid + s] == rv[tid] && ri[tid + s] < ri[tid])) {
                    rv[tid] = rv[tid + s];
                    ri[tid] = ri[tid + s];
                }
            }
            __syncthreads();
        }
        if (tid == 0) {
            int b = ri[0];
            if (b >= N_RES) b = 0;
            idxout[(size_t)i * KNN + it] = b;
            d2s[b] = 3.35e38f;
        }
        __syncthreads();
    }
}

// ---------------------------------------------------------------------------
// Kernel 4: edge features + GEMM1 (WMMA f16->f32) + LayerNorm + GEMM2 (WMMA)
// One block (256 thr = 8 waves) = 16 edge rows (one residue, 16 neighbors)
// x 128 output channels. Wave w owns columns [16w, 16w+16).
// ---------------------------------------------------------------------------
__global__ void pf_edges(const float* __restrict__ bb,
                         const int* __restrict__ idxbuf,
                         const int* __restrict__ resi,
                         const int* __restrict__ chain,
                         const float* __restrict__ w_pos_w,
                         const float* __restrict__ w_pos_b,
                         const _Float16* __restrict__ w_e_t,
                         const float* __restrict__ ln_w,
                         const float* __restrict__ ln_b,
                         const _Float16* __restrict__ proj_t,
                         const float* __restrict__ proj_b,
                         float* __restrict__ out) {
    __shared__ _Float16 feat[16 * DFEAT];
    __shared__ float    ebuf[16 * EDIM];
    __shared__ _Float16 anorm[16 * EDIM];
    __shared__ float    pdb[16 * 25];
    __shared__ float    bbi[15];
    __shared__ float    bbj[16 * 15];
    __shared__ int      jidx[16];
    __shared__ int      encv[16];
    __shared__ float    ps[256], ps2[256];
    __shared__ float    mmu[16], mri[16];

    const int tid = threadIdx.x;
    const int i  = blockIdx.x / 3;          // 16 | 48 => one residue per block
    const int k0 = (blockIdx.x % 3) * 16;
    const int g0 = blockIdx.x * 16;

    if (tid < 16) {
        int j = idxbuf[(size_t)i * KNN + k0 + tid];
        jidx[tid] = j;
        int off = resi[i] - resi[j] + MAXREL;
        off = off < 0 ? 0 : (off > 2 * MAXREL ? 2 * MAXREL : off);
        encv[tid] = (chain[i] == chain[j]) ? off : (2 * MAXREL + 1);
    }
    if (tid < 15) bbi[tid] = bb[(size_t)i * 15 + tid];
    __syncthreads();
    if (tid < 240) {
        int r = tid / 15, c = tid % 15;
        bbj[tid] = bb[(size_t)jidx[r] * 15 + c];
    }
    __syncthreads();
    if (tid < 400) {
        int r = tid / 25, p = tid % 25;
        int a = p / 5, b = p % 5;
        float dx = bbi[a * 3 + 0] - bbj[r * 15 + b * 3 + 0];
        float dy = bbi[a * 3 + 1] - bbj[r * 15 + b * 3 + 1];
        float dz = bbi[a * 3 + 2] - bbj[r * 15 + b * 3 + 2];
        pdb[tid] = sqrtf(dx * dx + dy * dy + dz * dz + 1e-6f);
    }
    __syncthreads();
    for (int f = tid; f < 16 * DFEAT; f += 256) {
        int r = f / DFEAT, d = f % DFEAT;
        float val;
        if (d < 16) {
            val = w_pos_w[d * 34 + encv[r]] + w_pos_b[d];
        } else {
            int q = d - 16;
            int p = q >> 4, t = q & 15;
            float mu = 2.0f + 20.0f * (float)t / 15.0f;   // linspace(2,22,16)
            float x = (pdb[r * 25 + p] - mu) * 0.8f;      // / (20/16)
            val = __expf(-x * x);
        }
        feat[r * DFEAT + d] = (_Float16)val;
    }
    __syncthreads();

    const int wave   = tid >> 5;
    const int lane   = tid & 31;
    const int m      = lane & 15;
    const int hiHalf = (lane & 16) ? 1 : 0;
    const int kaBase = hiHalf ? 16 : 0;
    const int col    = wave * 16 + m;

    const v16h* B1 = (const v16h*)w_e_t;     // swizzled fragments
    const v16h* B2 = (const v16h*)proj_t;

    // ---- GEMM1: e = edges(16x416) x W1(416x128), f16 in / f32 acc ----
    v8f acc = {};
    for (int kk = 0; kk < DFEAT / 32; ++kk) {
        v16h a;
#pragma unroll
        for (int h = 0; h < 16; ++h) {
            int ka = kk * 32 + h + kaBase;
            a[h] = feat[m * DFEAT + ka];
        }
        v16h b = B1[(kk * 8 + wave) * 32 + lane];   // one 32B vector load
        acc = __builtin_amdgcn_wmma_f32_16x16x32_f16(
            false, a, false, b, (short)0, acc, false, false);
    }
#pragma unroll
    for (int v = 0; v < 8; ++v) {
        int mrow = v + (hiHalf ? 8 : 0);
        ebuf[mrow * EDIM + col] = acc[v];
    }
    __syncthreads();

    // ---- LayerNorm over 128 channels (f32) ----
    {
        int r = tid >> 4, l = tid & 15;
        float s = 0.0f, s2 = 0.0f;
        for (int c = l; c < EDIM; c += 16) {
            float v = ebuf[r * EDIM + c];
            s += v; s2 += v * v;
        }
        ps[tid] = s; ps2[tid] = s2;
    }
    __syncthreads();
    if (tid < 16) {
        float s = 0.0f, s2 = 0.0f;
        for (int l = 0; l < 16; ++l) { s += ps[tid * 16 + l]; s2 += ps2[tid * 16 + l]; }
        float mean = s * (1.0f / EDIM);
        float var = s2 * (1.0f / EDIM) - mean * mean;
        mmu[tid] = mean;
        mri[tid] = rsqrtf(var + 1e-5f);
    }
    __syncthreads();
    for (int id = tid; id < 16 * EDIM; id += 256) {
        int r = id >> 7, c = id & 127;
        float v = (ebuf[id] - mmu[r]) * mri[r] * ln_w[c] + ln_b[c];
        anorm[id] = (_Float16)v;
    }
    __syncthreads();

    // ---- GEMM2: out = e_ln(16x128) x W2(128x128) + proj_b ----
    v8f acc2 = {};
    for (int kk = 0; kk < EDIM / 32; ++kk) {
        v16h a;
#pragma unroll
        for (int h = 0; h < 16; ++h) {
            int ka = kk * 32 + h + kaBase;
            a[h] = anorm[m * EDIM + ka];
        }
        v16h b = B2[(kk * 8 + wave) * 32 + lane];
        acc2 = __builtin_amdgcn_wmma_f32_16x16x32_f16(
            false, a, false, b, (short)0, acc2, false, false);
    }
    float pb = proj_b[col];
#pragma unroll
    for (int v = 0; v < 8; ++v) {
        int mrow = v + (hiHalf ? 8 : 0);
        out[(size_t)(g0 + mrow) * EDIM + col] = acc2[v] + pb;
    }
}

// ---------------------------------------------------------------------------
// Host launch
// ---------------------------------------------------------------------------
extern "C" void kernel_launch(void* const* d_in, const int* in_sizes, int n_in,
                              void* d_out, int out_size, void* d_ws, size_t ws_size,
                              hipStream_t stream) {
    const float* coords  = (const float*)d_in[0];
    const float* mask    = (const float*)d_in[1];
    const int*   resi    = (const int*)d_in[2];
    const int*   chain   = (const int*)d_in[3];
    const float* w_pos_w = (const float*)d_in[4];
    const float* w_pos_b = (const float*)d_in[5];
    const float* w_e_w   = (const float*)d_in[6];
    const float* ln_w    = (const float*)d_in[7];
    const float* ln_b    = (const float*)d_in[8];
    const float* proj_w  = (const float*)d_in[9];
    const float* proj_b  = (const float*)d_in[10];

    float* out = (float*)d_out;
    int* idxout = (int*)(out + (size_t)N_RES * KNN * EDIM);   // idx after out

    float* bb = (float*)d_ws;                                 // N*15 f32
    _Float16* w_e_t = (_Float16*)((char*)d_ws + (size_t)N_RES * 15 * sizeof(float));
    _Float16* proj_t = w_e_t + DFEAT * EDIM;

    pf_prep_bb<<<(N_RES + 255) / 256, 256, 0, stream>>>(coords, bb);
    int wtot = DFEAT * EDIM + EDIM * EDIM;
    pf_prep_w<<<(wtot + 255) / 256, 256, 0, stream>>>(w_e_w, proj_w, w_e_t, proj_t);
    pf_topk<<<N_RES, 256, 0, stream>>>(coords, mask, idxout);
    pf_edges<<<(N_RES * KNN) / 16, 256, 0, stream>>>(
        bb, idxout, resi, chain, w_pos_w, w_pos_b, w_e_t,
        ln_w, ln_b, proj_t, proj_b, out);
}
